// PerElementModel_39333310496837
// MI455X (gfx1250) — matrix operand order repaired and checked by reading, not compiled, not run on hardware
//
#include <hip/hip_runtime.h>

// GPR inference, MI455X (gfx1250, wave32, WMMA).
// - In-block compaction by element => 1x (not 4x) exp/GEMM work.
// - f16 WMMA 16x16x32: K=32 == N_DESC, one WMMA per 16-atom x 16-point tile.
// - A,B prescaled by sqrt(2*log2e) and the -log2e*(x2+z2) bias is fed through
//   the WMMA C operand, so D = -log2e*dist directly; inner loop is then just
//   raw v_exp_f32 + fma-accumulate.
// - Exps computed into distinct temps to break the exp->fmac serial chain.

typedef __attribute__((ext_vector_type(16))) _Float16 v16h;
typedef __attribute__((ext_vector_type(8)))  float    v8f;

#define NMODELS 4
#define NPOINTS 512
#define NDESC   32
#define BLOCK_ATOMS 1024
#define THREADS 256          // 8 waves (wave32)
#define PT_TILES (NPOINTS / 16)
#define L2E      1.4426950408889634f
#define SQ2L2E   1.69864362f   // sqrt(2*log2e); S*S = 2*log2e

__global__ __launch_bounds__(THREADS)
void gpr_routed_kernel(const int*   __restrict__ element,
                       const float* __restrict__ x,       // [N, 32]
                       const float* __restrict__ Z,       // [4, 512, 32]
                       const float* __restrict__ alpha,   // [4, 512]
                       const float* __restrict__ logls,   // [4, 32]
                       float*       __restrict__ out,     // [N]
                       int n_atoms)
{
    __shared__ alignas(32) _Float16 sB[NPOINTS * NDESC]; // S-scaled Z, f16, point-major (32 KB)
    __shared__ float2 sAZ[NPOINTS];                      // {alpha, -log2e*z2}
    __shared__ float  sIls[NDESC];
    __shared__ unsigned short sIdx[BLOCK_ATOMS];
    __shared__ int sCnt[NMODELS], sOff[NMODELS], sFill[NMODELS];

    const int tid  = threadIdx.x;
    const int lane = tid & 31;
    const int wave = tid >> 5;
    const int hi   = lane >> 4;        // half-wave select
    const int rlo  = lane & 15;
    const int blockStart = blockIdx.x * BLOCK_ATOMS;

    // ---------- Phase 0: compact atoms by element into per-model lists ----------
    if (tid < NMODELS) sCnt[tid] = 0;
    __syncthreads();

    int e0[BLOCK_ATOMS / THREADS];
#pragma unroll
    for (int k = 0; k < BLOCK_ATOMS / THREADS; ++k) {
        int la = k * THREADS + tid;
        int a  = blockStart + la;
        int e  = (a < n_atoms) ? element[a] : -1;
        e0[k] = e;
        if (e >= 0) atomicAdd(&sCnt[e], 1);
    }
    __syncthreads();
    if (tid == 0) {
        int o = 0;
        for (int m = 0; m < NMODELS; ++m) { sOff[m] = o; sFill[m] = o; o += sCnt[m]; }
    }
    __syncthreads();
#pragma unroll
    for (int k = 0; k < BLOCK_ATOMS / THREADS; ++k) {
        if (e0[k] >= 0) {
            int pos = atomicAdd(&sFill[e0[k]], 1);
            sIdx[pos] = (unsigned short)(k * THREADS + tid);
        }
    }

    // ---------- Per-model passes ----------
    for (int m = 0; m < NMODELS; ++m) {
        __syncthreads();
        if (tid < NDESC) sIls[tid] = __expf(-0.5f * logls[m * NDESC + tid]);
        __syncthreads();

        // Stage S-scaled inducing points as f16 (point-major => B fragment is
        // 16 contiguous halfs per lane: point = pt*16 + (lane&15), K half = hi*16).
        const float* Zm = Z + (size_t)m * NPOINTS * NDESC;
        for (int i = tid; i < NPOINTS * NDESC; i += THREADS) {
            float v = Zm[i] * sIls[i & (NDESC - 1)] * SQ2L2E;
            sB[i] = (_Float16)v;
        }
        __syncthreads();
        // {alpha, -log2e*z2}: since sB is scaled by S with S^2 = 2*log2e,
        // -log2e*z2_true = -0.5 * sum(sB^2).
        for (int p = tid; p < NPOINTS; p += THREADS) {
            float s = 0.f;
#pragma unroll
            for (int d = 0; d < NDESC; ++d) {
                float v = (float)sB[p * NDESC + d];
                s += v * v;
            }
            sAZ[p] = make_float2(alpha[m * NPOINTS + p], -0.5f * s);
        }
        __syncthreads();

        const int cnt = sCnt[m];
        const int off = sOff[m];
        const int ntiles = (cnt + 15) >> 4;

        for (int t = wave; t < ntiles; t += THREADS / 32) {
            // ---- build A fragment: row = lane&15, K = {hi*16..+15}
            int pos  = t * 16 + rlo;
            int lidx = sIdx[off + ((pos < cnt) ? pos : 0)];   // pad rows duplicate row 0
            const float* xr = x + (size_t)(blockStart + lidx) * NDESC;
            const int kbase = hi * 16;

            float xb[16];
            *(float4*)&xb[0]  = *(const float4*)(xr + kbase);
            *(float4*)&xb[4]  = *(const float4*)(xr + kbase + 4);
            *(float4*)&xb[8]  = *(const float4*)(xr + kbase + 8);
            *(float4*)&xb[12] = *(const float4*)(xr + kbase + 12);

            v16h afrag;
            float part = 0.f;
#pragma unroll
            for (int i = 0; i < 8; ++i) {
                float v = xb[i] * sIls[kbase + i];
                part += v * v;
                afrag[i] = (_Float16)(v * SQ2L2E);
            }
#pragma unroll
            for (int i = 0; i < 8; ++i) {
                float v = xb[8 + i] * sIls[kbase + 8 + i];
                part += v * v;
                afrag[8 + i] = (_Float16)(v * SQ2L2E);
            }
            // full ||x||^2 for row (lane&15): combine the two K-halves
            float x2 = part + __shfl_xor(part, 16);
            // per-lane bias for its 8 C rows (rows v + 8*hi): -log2e * x2[row]
            float q[8];
#pragma unroll
            for (int v = 0; v < 8; ++v)
                q[v] = -L2E * __shfl(x2, v + 8 * hi);

            float acc[8] = {0.f, 0.f, 0.f, 0.f, 0.f, 0.f, 0.f, 0.f};

#pragma unroll 2
            for (int pt = 0; pt < PT_TILES; ++pt) {
                int col = pt * 16 + rlo;
                // B fragment: 16 contiguous halfs (32B-aligned) from LDS
                v16h bfrag = *(const v16h*)(sB + col * NDESC + hi * 16);
                float2 az = sAZ[col];
                float  al = az.x;
                float  qb = az.y;

                // C-in = -log2e*(x2_row + z2_col); WMMA adds 2*log2e*cross
                v8f cin;
#pragma unroll
                for (int v = 0; v < 8; ++v) cin[v] = q[v] + qb;

                v8f d = __builtin_amdgcn_wmma_f32_16x16x32_f16(
                            false, afrag, false, bfrag, (short)0, cin, false, false);

                // distinct temps: exps pipeline at trans throughput, fmacs can pack
                float e[8];
#pragma unroll
                for (int v = 0; v < 8; ++v)
                    e[v] = __builtin_amdgcn_exp2f(d[v]);
#pragma unroll
                for (int v = 0; v < 8; ++v)
                    acc[v] = fmaf(al, e[v], acc[v]);
            }

            // reduce over the 16 columns held by each half-wave, then scatter
#pragma unroll
            for (int v = 0; v < 8; ++v) {
                float s = acc[v];
                s += __shfl_xor(s, 1);
                s += __shfl_xor(s, 2);
                s += __shfl_xor(s, 4);
                s += __shfl_xor(s, 8);
                if (rlo == v) {
                    int row = v + 8 * hi;
                    int p2  = t * 16 + row;
                    if (p2 < cnt) {
                        int a2 = blockStart + sIdx[off + p2];
                        out[a2] = s;
                    }
                }
            }
        }
    }
}

extern "C" void kernel_launch(void* const* d_in, const int* in_sizes, int n_in,
                              void* d_out, int out_size, void* d_ws, size_t ws_size,
                              hipStream_t stream) {
    (void)n_in; (void)d_ws; (void)ws_size; (void)out_size;
    const int*   element = (const int*)d_in[0];
    const float* x       = (const float*)d_in[1];
    const float* Z       = (const float*)d_in[2];
    const float* alpha   = (const float*)d_in[3];
    const float* logls   = (const float*)d_in[4];
    float*       out     = (float*)d_out;

    int n = in_sizes[0];
    int nblocks = (n + BLOCK_ATOMS - 1) / BLOCK_ATOMS;
    gpr_routed_kernel<<<nblocks, THREADS, 0, stream>>>(
        element, x, Z, alpha, logls, out, n);
}
